// LSTMRnn_111669150174
// MI455X (gfx1250) — compile-verified, matrix operand order checked by reading, not run on hardware
//
#include <hip/hip_runtime.h>
#include <hip/hip_bf16.h>
#include <stdint.h>

// ---------------------------------------------------------------------------
// LSTM persistent kernel for MI455X (gfx1250), wave32 + WMMA bf16.
//   grid  = 8 blocks (one per 16-row batch tile)
//   block = 512 threads = 16 waves
//   wave w owns hidden columns [16w, 16w+16): computes gate tiles i,f,o,cg
//   h state: LDS (double-buffered), bf16, WMMA A-operand layout
//   c state: registers (v8f per wave = 16x16 f32 C-tile)
//   Uh     : bf16, pre-swizzled B-operand tiles, L2-resident, re-streamed
//            each step (opaque *offset* defeats LICM but keeps the global
//            address space so loads lower to global_load_b128 / LOADcnt)
// ---------------------------------------------------------------------------

typedef __bf16 v16bf __attribute__((ext_vector_type(16)));
typedef float  v8f   __attribute__((ext_vector_type(8)));

#define TPB     512
#define T_STEPS 1024
#define P_STEPS 8
#define HID     256
#define G4      1024
#define BTILE   16
#define KTILES  8      // 256 / 32
#define NTILES  64     // 1024 / 16

__device__ __forceinline__ float sigmoid_f(float x) {
    return 1.0f / (1.0f + __expf(-x));
}

// branch-free tanh on the fast exp path; saturates to +/-1 correctly
__device__ __forceinline__ float tanh_f(float x) {
    float e = __expf(2.0f * x);
    return 1.0f - 2.0f / (e + 1.0f);
}

__device__ __forceinline__ unsigned short f32_to_bf16(float f) {
    uint32_t b = __float_as_uint(f);
    return (unsigned short)((b + 0x7FFFu + ((b >> 16) & 1u)) >> 16);
}

// ---------------------------------------------------------------------------
// Pre-pass: Uh [256,1024] f32 -> bf16 WMMA-B tiles.
// Tile (kk, nt): 32x16 (KxN) block, stored as 512 bf16, lane-major:
//   element (lane l, i): N = nt*16 + l%16,
//   K_i = (i<8) ? 8*(l/16)+i : 16 + 8*(l/16) + (i-8)
// so the hot loop loads one contiguous v16bf (32B) per lane per tile.
// ---------------------------------------------------------------------------
__global__ void lstm_convert_uh(const float* __restrict__ Uh,
                                unsigned short* __restrict__ dst) {
    int gid  = blockIdx.x * blockDim.x + threadIdx.x;   // 0 .. 262143
    int e    = gid & 511;
    int tile = gid >> 9;
    int kk   = tile >> 6;
    int nt   = tile & 63;
    int l    = e >> 4;
    int i    = e & 15;
    int h2   = l >> 4;
    int nl   = l & 15;
    int Ki   = (i < 8) ? (8 * h2 + i) : (16 + 8 * h2 + (i - 8));
    int K    = kk * 32 + Ki;
    int N    = nt * 16 + nl;
    dst[gid] = f32_to_bf16(Uh[K * G4 + N]);
}

// ---------------------------------------------------------------------------
// Main persistent LSTM kernel.
// ---------------------------------------------------------------------------
__global__ void __launch_bounds__(TPB, 1)
lstm_persistent(const float* __restrict__ seq,    // [128,1024]
                const float* __restrict__ Wx,     // [1,1024]
                const float* __restrict__ bu,     // [1024]
                const float* __restrict__ Wout,   // [256,1]
                const float* __restrict__ bout,   // [1]
                const unsigned short* __restrict__ uh_b, // swizzled bf16 tiles
                float* __restrict__ out)          // [128, 1032]
{
    // h double buffer in WMMA A-operand layout (removes mid-step barrier)
    __shared__ __align__(32) unsigned short h_lds[2][KTILES * 512]; // 16KB
    __shared__ float wx_s[G4];
    __shared__ float bu_s[G4];
    __shared__ float wout_s[HID];
    __shared__ float out_acc[2][BTILE];   // double-buffered output reduction
    __shared__ float s_x[2][BTILE];       // autoregressive feedback input

    const int tid   = threadIdx.x;
    const int b0    = blockIdx.x * BTILE;
    const int w     = tid >> 5;          // wave id, 0..15
    const int lane  = tid & 31;
    const int lhalf = lane >> 4;
    const int lmod  = lane & 15;

    // --- one-time staging -------------------------------------------------
    for (int i = tid; i < G4; i += TPB) { wx_s[i] = Wx[i]; bu_s[i] = bu[i]; }
    for (int i = tid; i < HID; i += TPB) wout_s[i] = Wout[i];
    for (int i = tid; i < KTILES * 512; i += TPB) {
        h_lds[0][i] = 0;  h_lds[1][i] = 0;      // h0 = 0
    }
    if (tid < BTILE) {
        out_acc[0][tid] = 0.0f; out_acc[1][tid] = 0.0f;
        s_x[0][tid] = 0.0f;     s_x[1][tid] = 0.0f;
    }
    const float boutv = bout[0];
    __syncthreads();

    // --- per-lane constants ----------------------------------------------
    const int jcol = 16 * w + lmod;               // hidden column 0..255
    float wx_g[4], bu_g[4];
    #pragma unroll
    for (int g = 0; g < 4; ++g) {
        int n = 256 * g + jcol;
        wx_g[g] = wx_s[n];
        bu_g[g] = bu_s[n];
    }
    const float woutj = wout_s[jcol];

    // h write-back address (A-operand layout), per VGPR r: hw_base + 16*r
    const int Kl  = jcol & 31;
    const int h2w = (Kl < 16) ? (Kl >> 3) : ((Kl - 16) >> 3);
    const int iw  = (Kl < 16) ? (Kl & 7) : (8 + (Kl & 7));
    const int kkw = jcol >> 5;
    const int hw_base = kkw * 512 + (8 * lhalf + 16 * h2w) * 16 + iw;

    v8f cst = {};   // cell state tile (16x16 f32 striped across wave)

    // --- recurrent loop ---------------------------------------------------
    for (int t = 0; t < T_STEPS + P_STEPS; ++t) {
        const int rb = t & 1;          // h read buffer / out+x buffer
        const int wb = rb ^ 1;         // h write buffer

        // Opaque zero offset: defeats LICM (B tiles are re-streamed from the
        // L2-resident copy every step instead of hoisted + spilled), while
        // keeping uh_b's provenance so address-space inference still emits
        // global_load_b128 (LOADcnt only; DScnt stays free for the A tiles).
        uint32_t uoff = 0;
        asm volatile("" : "+s"(uoff));
        const unsigned short* ubp = uh_b + uoff;

        // gate init: g = x*Wx + bu  (IN==1 -> scalar * row of Wx)
        float xr[8];
        if (t < T_STEPS) {
            #pragma unroll
            for (int r = 0; r < 8; ++r)
                xr[r] = seq[(b0 + r + 8 * lhalf) * T_STEPS + t]; // L2 broadcast
        } else {
            #pragma unroll
            for (int r = 0; r < 8; ++r)
                xr[r] = s_x[rb][r + 8 * lhalf];
        }

        v8f acc[4];
        #pragma unroll
        for (int g = 0; g < 4; ++g) {
            #pragma unroll
            for (int r = 0; r < 8; ++r)
                acc[g][r] = xr[r] * wx_g[g] + bu_g[g];
        }

        // g += h @ Uh : 8 K-slices x 4 gate tiles, software-pipelined B
        v16bf bcur[4];
        #pragma unroll
        for (int g = 0; g < 4; ++g)
            bcur[g] = *(const v16bf*)(ubp + ((w + 16 * g) * 512 + lane * 16));

        #pragma unroll
        for (int kk = 0; kk < KTILES; ++kk) {
            v16bf a = *(const v16bf*)(&h_lds[rb][kk * 512 + lane * 16]);
            v16bf bnxt[4];
            if (kk < KTILES - 1) {
                #pragma unroll
                for (int g = 0; g < 4; ++g)
                    bnxt[g] = *(const v16bf*)(ubp + (((kk + 1) * 64 + w + 16 * g) * 512
                                                     + lane * 16));
            }
            #pragma unroll
            for (int g = 0; g < 4; ++g) {
                acc[g] = __builtin_amdgcn_wmma_f32_16x16x32_bf16(
                    false, a, false, bcur[g], (short)0, acc[g], false, false);
            }
            if (kk < KTILES - 1) {
                #pragma unroll
                for (int g = 0; g < 4; ++g) bcur[g] = bnxt[g];
            }
        }

        // cell update + h write-back (other buffer) + output partials
        float pr[8];
        #pragma unroll
        for (int r = 0; r < 8; ++r) {
            float it = sigmoid_f(acc[0][r]);
            float ft = sigmoid_f(acc[1][r]);
            float ot = sigmoid_f(acc[2][r]);
            float ct = ft * cst[r] + it * tanh_f(acc[3][r]);
            cst[r] = ct;
            float ht = ot * tanh_f(ct);
            h_lds[wb][hw_base + 16 * r] = f32_to_bf16(ht);
            pr[r] = ht * woutj;
        }

        // o = h @ Wout : reduce each row across the 16 lanes holding it
        #pragma unroll
        for (int r = 0; r < 8; ++r) {
            float v = pr[r];
            v += __shfl_xor(v, 1);
            v += __shfl_xor(v, 2);
            v += __shfl_xor(v, 4);
            v += __shfl_xor(v, 8);
            if (lmod == 0) atomicAdd(&out_acc[rb][r + 8 * lhalf], v);
        }

        __syncthreads();   // single per-step barrier (h + out reduction)

        if (tid < BTILE) {
            float o = out_acc[rb][tid] + boutv;
            out[(b0 + tid) * (T_STEPS + P_STEPS) + t] = o;
            out_acc[rb][tid] = 0.0f;       // reused at step t+2 (after B_{t+1})
            s_x[wb][tid] = o;              // feedback input for step t+1
        }
        // Only the autoregressive steps consume s_x next step, so only they
        // need the write->read fence; teacher-forced steps run 1 barrier/step.
        if (t >= T_STEPS - 1) __syncthreads();
    }
}

// ---------------------------------------------------------------------------
extern "C" void kernel_launch(void* const* d_in, const int* in_sizes, int n_in,
                              void* d_out, int out_size, void* d_ws, size_t ws_size,
                              hipStream_t stream) {
    (void)in_sizes; (void)n_in; (void)out_size; (void)ws_size;
    const float* seq  = (const float*)d_in[0];   // [128,1024]
    const float* Wx   = (const float*)d_in[1];   // [1,1024]
    const float* Uh   = (const float*)d_in[2];   // [256,1024]
    const float* bu   = (const float*)d_in[3];   // [1024]
    const float* Wout = (const float*)d_in[4];   // [256,1]
    const float* bout = (const float*)d_in[5];   // [1]
    float* out = (float*)d_out;                  // [128,1032]

    unsigned short* uh_b = (unsigned short*)d_ws;  // 512KB bf16 swizzled Uh

    // Pre-pass: convert + swizzle Uh into WMMA B-tile layout (L2-resident).
    lstm_convert_uh<<<(512 * 512) / 256, 256, 0, stream>>>(Uh, uh_b);

    // Persistent recurrent kernel: 8 batch tiles x 16 waves.
    lstm_persistent<<<8, TPB, 0, stream>>>(seq, Wx, bu, Wout, bout, uh_b, out);
}
